// LiveNet_7413113553682
// MI455X (gfx1250) — matrix operand order repaired and checked by reading, not compiled, not run on hardware
//
#include <hip/hip_runtime.h>
#include <stdint.h>

typedef __attribute__((ext_vector_type(16))) __bf16 v16bf;
typedef __attribute__((ext_vector_type(8)))  __bf16 v8bf;
typedef __attribute__((ext_vector_type(4)))  __bf16 v4bf;
typedef __attribute__((ext_vector_type(8)))  float  v8f;
typedef __attribute__((ext_vector_type(4)))  float  f32x4;

// Block = 256 threads = 8 waves (wave32). Block tile 128x128, k-step 32.
// Wave grid 2(M) x 4(N); wave tile 64x32 = 4x2 WMMA accumulators.
#define BM  128
#define BN  128
#define BK  32
#define LDT 40                     // padded LDS row: 40 bf16 = 80B (16B aligned)
#define PLANE_B (BM * LDT * 2)     // 10240 bytes per plane
#define BUF_B   (4 * PLANE_B)      // Ah, Al, Bh, Bl = 40960 bytes per buffer

#define USE_ASYNC_COPY 1           // gfx1250 global_load_async_to_lds_b128 path

__device__ __forceinline__ void split_bf16(float x, __bf16& hi, __bf16& lo) {
    hi = (__bf16)x;                 // RNE round to bf16
    lo = (__bf16)(x - (float)hi);   // residual (next ~8 mantissa bits)
}

// ---------------------------------------------------------------------------
// Pre-pass 1: elementwise fp32 -> bf16 hi/lo planes (for x)
// ---------------------------------------------------------------------------
__global__ __launch_bounds__(256)
void split_plane(const float* __restrict__ in, __bf16* __restrict__ hi,
                 __bf16* __restrict__ lo, int n) {
    int i = (blockIdx.x * 256 + threadIdx.x) * 4;
    if (i >= n) return;
    f32x4 v = *(const f32x4*)(in + i);
    v4bf h, l;
#pragma unroll
    for (int j = 0; j < 4; ++j) { __bf16 hh, ll; split_bf16(v[j], hh, ll); h[j] = hh; l[j] = ll; }
    *(v4bf*)(hi + i) = h;
    *(v4bf*)(lo + i) = l;
}

// ---------------------------------------------------------------------------
// Pre-pass 2: W [K][N] fp32 -> transposed bf16 hi/lo planes [N][K]
// ---------------------------------------------------------------------------
__global__ __launch_bounds__(256)
void split_transpose(const float* __restrict__ W, __bf16* __restrict__ Th,
                     __bf16* __restrict__ Tl, int K, int N) {
    __shared__ __bf16 th[32][33];
    __shared__ __bf16 tl[32][33];
    const int k0 = blockIdx.y * 32, n0 = blockIdx.x * 32;
    const int tx = threadIdx.x & 31, ty = threadIdx.x >> 5;
#pragma unroll
    for (int i = 0; i < 4; ++i) {
        const int k = ty + i * 8;
        float v = W[(size_t)(k0 + k) * N + n0 + tx];
        __bf16 h, l; split_bf16(v, h, l);
        th[k][tx] = h; tl[k][tx] = l;
    }
    __syncthreads();
#pragma unroll
    for (int i = 0; i < 4; ++i) {
        const int nn = ty + i * 8;
        Th[(size_t)(n0 + nn) * K + k0 + tx] = th[tx][nn];
        Tl[(size_t)(n0 + nn) * K + k0 + tx] = tl[tx][nn];
    }
}

// ---------------------------------------------------------------------------
// Async global->LDS copy (gfx1250 Tensor/async path, ASYNCcnt-tracked)
// ---------------------------------------------------------------------------
__device__ __forceinline__ void async_b128(uint32_t lds_off, const void* gptr) {
#if USE_ASYNC_COPY
    asm volatile("global_load_async_to_lds_b128 %0, %1, off"
                 :: "v"(lds_off), "v"(gptr) : "memory");
#endif
}
__device__ __forceinline__ void wait_async_all() {
#if USE_ASYNC_COPY
    asm volatile("s_wait_asynccnt 0x0" ::: "memory");
#endif
}

// ---------------------------------------------------------------------------
// GEMM on pre-split bf16 planes: C = A(MxK) * B^T(NxK) + bias, bf16x3 scheme.
// SPLIT_OUT: write result as bf16 hi/lo planes (for h); else fp32.
// ---------------------------------------------------------------------------
template<bool RELU, bool SPLIT_OUT>
__global__ __launch_bounds__(256)
void gemm_planes(const __bf16* __restrict__ Ah, const __bf16* __restrict__ Al,
                 const __bf16* __restrict__ Bh, const __bf16* __restrict__ Bl,
                 const float*  __restrict__ bias,
                 float* __restrict__ C,
                 __bf16* __restrict__ Oh, __bf16* __restrict__ Ol,
                 int M, int N, int K)
{
    __shared__ __align__(16) uint8_t smem[2 * BUF_B];   // double buffered

    const int t    = threadIdx.x;
    const int lane = t & 31;
    const int wid  = t >> 5;
    const int wm   = wid >> 2;     // 0..1
    const int wn   = wid & 3;      // 0..3
    const int g    = lane >> 4;    // lane group
    const int l16  = lane & 15;

    const int m0 = blockIdx.y * BM;
    const int n0 = blockIdx.x * BN;
    (void)M;

    // Staging map: 512 16B chunks per plane; thread does chunks t and t+256.
    const int q0_r = t >> 2,          q0_c = t & 3;
    const int q1_r = (t + 256) >> 2,  q1_c = t & 3;   // (t+256)&3 == t&3

    v8f acc[4][2];
#pragma unroll
    for (int i = 0; i < 4; ++i)
#pragma unroll
        for (int j = 0; j < 2; ++j)
            acc[i][j] = (v8f){0.f, 0.f, 0.f, 0.f, 0.f, 0.f, 0.f, 0.f};

    auto stage = [&](int buf, int k0) {
        const uint32_t base = (uint32_t)(buf * BUF_B);
#if !USE_ASYNC_COPY
        __bf16* dst = (__bf16*)(smem + base);
#endif
#pragma unroll
        for (int i = 0; i < 2; ++i) {
            const int r = (i == 0) ? q0_r : q1_r;
            const int c = (i == 0) ? q0_c : q1_c;
            const uint32_t loff = (uint32_t)(r * (LDT * 2) + c * 16);
            const size_t ga = (size_t)(m0 + r) * K + k0 + c * 8;
            const size_t gb = (size_t)(n0 + r) * K + k0 + c * 8;
#if USE_ASYNC_COPY
            async_b128(base + 0 * PLANE_B + loff, Ah + ga);
            async_b128(base + 1 * PLANE_B + loff, Al + ga);
            async_b128(base + 2 * PLANE_B + loff, Bh + gb);
            async_b128(base + 3 * PLANE_B + loff, Bl + gb);
#else
            *(v8bf*)((uint8_t*)dst + 0 * PLANE_B + loff) = *(const v8bf*)(Ah + ga);
            *(v8bf*)((uint8_t*)dst + 1 * PLANE_B + loff) = *(const v8bf*)(Al + ga);
            *(v8bf*)((uint8_t*)dst + 2 * PLANE_B + loff) = *(const v8bf*)(Bh + gb);
            *(v8bf*)((uint8_t*)dst + 3 * PLANE_B + loff) = *(const v8bf*)(Bl + gb);
#endif
        }
    };

    // Prologue: fill buffer 0.
    stage(0, 0);

    int cur = 0;
    for (int k0 = 0; k0 < K; k0 += BK) {
        wait_async_all();        // current buffer's async copies have landed
        __syncthreads();         // ...for every wave in the block

        const __bf16* sAh = (const __bf16*)(smem + cur * BUF_B + 0 * PLANE_B);
        const __bf16* sAl = (const __bf16*)(smem + cur * BUF_B + 1 * PLANE_B);
        const __bf16* sBh = (const __bf16*)(smem + cur * BUF_B + 2 * PLANE_B);
        const __bf16* sBl = (const __bf16*)(smem + cur * BUF_B + 3 * PLANE_B);

        // Load ALL fragments for this k-step before restaging (ds reads are
        // dscnt-waited before the WMMAs, which precedes the next barrier).
        v16bf fah[4], fal[4], fbh[2], fbl[2];
#pragma unroll
        for (int tm = 0; tm < 4; ++tm) {
            const int r = wm * 64 + tm * 16 + l16;
            v8bf h0 = *(const v8bf*)(sAh + r * LDT + 8 * g);
            v8bf h1 = *(const v8bf*)(sAh + r * LDT + 16 + 8 * g);
            fah[tm] = __builtin_shufflevector(h0, h1, 0,1,2,3,4,5,6,7,8,9,10,11,12,13,14,15);
            v8bf l0 = *(const v8bf*)(sAl + r * LDT + 8 * g);
            v8bf l1 = *(const v8bf*)(sAl + r * LDT + 16 + 8 * g);
            fal[tm] = __builtin_shufflevector(l0, l1, 0,1,2,3,4,5,6,7,8,9,10,11,12,13,14,15);
        }
#pragma unroll
        for (int tn = 0; tn < 2; ++tn) {
            const int c = wn * 32 + tn * 16 + l16;
            v8bf h0 = *(const v8bf*)(sBh + c * LDT + 8 * g);
            v8bf h1 = *(const v8bf*)(sBh + c * LDT + 16 + 8 * g);
            fbh[tn] = __builtin_shufflevector(h0, h1, 0,1,2,3,4,5,6,7,8,9,10,11,12,13,14,15);
            v8bf l0 = *(const v8bf*)(sBl + c * LDT + 8 * g);
            v8bf l1 = *(const v8bf*)(sBl + c * LDT + 16 + 8 * g);
            fbl[tn] = __builtin_shufflevector(l0, l1, 0,1,2,3,4,5,6,7,8,9,10,11,12,13,14,15);
        }

        // Kick off next tile's DMA; it overlaps with the WMMAs below.
        if (k0 + BK < K) stage(cur ^ 1, k0 + BK);

        // bf16x3 compensated product: Ah*Bh + Ah*Bl + Al*Bh
#pragma unroll
        for (int tm = 0; tm < 4; ++tm) {
#pragma unroll
            for (int tn = 0; tn < 2; ++tn) {
                acc[tm][tn] = __builtin_amdgcn_wmma_f32_16x16x32_bf16(
                    false, fah[tm], false, fbh[tn], (short)0, acc[tm][tn], false, false);
                acc[tm][tn] = __builtin_amdgcn_wmma_f32_16x16x32_bf16(
                    false, fah[tm], false, fbl[tn], (short)0, acc[tm][tn], false, false);
                acc[tm][tn] = __builtin_amdgcn_wmma_f32_16x16x32_bf16(
                    false, fal[tm], false, fbh[tn], (short)0, acc[tm][tn], false, false);
            }
        }
        cur ^= 1;
    }

    // Epilogue: bias (+ReLU); C/D layout: VGPR r -> M = 8g + r, N = l16.
#pragma unroll
    for (int tn = 0; tn < 2; ++tn) {
        const int col = n0 + wn * 32 + tn * 16 + l16;
        const float bv = bias[col];
#pragma unroll
        for (int tm = 0; tm < 4; ++tm) {
#pragma unroll
            for (int r = 0; r < 8; ++r) {
                const int row = m0 + wm * 64 + tm * 16 + 8 * g + r;
                float y = acc[tm][tn][r] + bv;
                if (RELU) y = fmaxf(y, 0.f);
                const size_t idx = (size_t)row * N + col;
                if (SPLIT_OUT) {
                    __bf16 hh, ll; split_bf16(y, hh, ll);
                    Oh[idx] = hh; Ol[idx] = ll;
                } else {
                    C[idx] = y;
                }
            }
        }
    }
}

extern "C" void kernel_launch(void* const* d_in, const int* in_sizes, int n_in,
                              void* d_out, int out_size, void* d_ws, size_t ws_size,
                              hipStream_t stream) {
    const float* x  = (const float*)d_in[0];   // [4096,1024]
    const float* W1 = (const float*)d_in[1];   // [1024,4096]
    const float* b1 = (const float*)d_in[2];   // [4096]
    const float* W2 = (const float*)d_in[3];   // [4096,1024]
    const float* b2 = (const float*)d_in[4];   // [1024]
    float* out = (float*)d_out;                // [4096,1024]
    (void)in_sizes; (void)n_in; (void)out_size; (void)ws_size;

    constexpr int Bsz = 4096, NIN = 1024, NHID = 4096, NOUT = 1024;
    constexpr size_t MB = 1u << 20;

    uint8_t* ws = (uint8_t*)d_ws;              // 112 MB layout:
    __bf16* h_hi  = (__bf16*)(ws +   0 * MB);  // [4096,4096] 32 MB
    __bf16* h_lo  = (__bf16*)(ws +  32 * MB);  // 32 MB
    __bf16* x_hi  = (__bf16*)(ws +  64 * MB);  // [4096,1024]  8 MB
    __bf16* x_lo  = (__bf16*)(ws +  72 * MB);
    __bf16* w1_hi = (__bf16*)(ws +  80 * MB);  // W1^T [4096,1024] 8 MB
    __bf16* w1_lo = (__bf16*)(ws +  88 * MB);
    __bf16* w2_hi = (__bf16*)(ws +  96 * MB);  // W2^T [1024,4096] 8 MB
    __bf16* w2_lo = (__bf16*)(ws + 104 * MB);

    dim3 blk(256);

    // Pre-split (and pre-transpose weights) once per launch.
    split_plane<<<(Bsz * NIN) / 1024, blk, 0, stream>>>(x, x_hi, x_lo, Bsz * NIN);
    split_transpose<<<dim3(NHID / 32, NIN  / 32), blk, 0, stream>>>(W1, w1_hi, w1_lo, NIN,  NHID);
    split_transpose<<<dim3(NOUT / 32, NHID / 32), blk, 0, stream>>>(W2, w2_hi, w2_lo, NHID, NOUT);

    // Layer 1: h = relu(x @ W1 + b1), written as bf16 hi/lo planes.
    gemm_planes<true, true><<<dim3(NHID / BN, Bsz / BM), blk, 0, stream>>>(
        x_hi, x_lo, w1_hi, w1_lo, b1, nullptr, h_hi, h_lo, Bsz, NHID, NIN);

    // Layer 2: y = h @ W2 + b2, fp32 output.
    gemm_planes<false, false><<<dim3(NOUT / BN, Bsz / BM), blk, 0, stream>>>(
        h_hi, h_lo, w2_hi, w2_lo, b2, out, nullptr, nullptr, Bsz, NOUT, NHID);
}
